// CenterSurroundConvolution_37460704755913
// MI455X (gfx1250) — matrix operand order repaired and checked by reading, not compiled, not run on hardware
//
#include <hip/hip_runtime.h>

// CenterSurroundConvolution as a K=512 bf16 WMMA GEMM (f32 accumulate).
// out = center @ (w_c - w_s) + box @ w_s + bias
// Roofline: 66.6 GFLOP vs 0.53 GB moved -> bf16 matrix pipe balances 23.3 TB/s HBM.

typedef __attribute__((ext_vector_type(16))) __bf16 v16bf;
typedef __attribute__((ext_vector_type(8)))  float  v8f;
typedef __attribute__((ext_vector_type(2)))  float  v2f;
typedef __attribute__((ext_vector_type(4)))  unsigned int uint4v;

#define B_    16
#define CIN   256
#define COUT  256
#define H_    128
#define W_    128
#define HO    126
#define WO    126

#define TILE_W 32
#define TILE_H 4
#define MT    (TILE_W * TILE_H)   // 128 spatial positions per workgroup
#define KTOT  512                 // 256 center + 256 box
#define LDS_K 520                 // +16B row padding: lanes hit distinct bank groups

// ---------------------------------------------------------------------------
// Prep: pack [w_c - w_s ; w_s] (fp32, 512x256) into this kernel's contiguous-K bf16 layout:
// Bp[(k>>4)*256 + n][k&15] -> a lane's 16 consecutive K values for column n
// are 32 contiguous bytes (two global_load_b128). Total 256 KB, L2-resident.
// ---------------------------------------------------------------------------
__global__ void prep_weights(const float* __restrict__ w_c,
                             const float* __restrict__ w_s,
                             __bf16* __restrict__ Bp) {
    int idx = blockIdx.x * blockDim.x + threadIdx.x;   // 0 .. 512*256-1
    int k = idx >> 8;        // 0..511
    int n = idx & 255;       // 0..255
    float v = (k < CIN) ? (w_c[k * COUT + n] - w_s[k * COUT + n])
                        : w_s[(k - CIN) * COUT + n];
    Bp[(((k >> 4) * COUT) + n) * 16 + (k & 15)] = (__bf16)v;
}

// ---------------------------------------------------------------------------
// Main fused kernel: 512 threads = 16 wave32; each wave owns a 16-wide slice
// of C_OUT. Workgroup tile: 32(w) x 4(h) positions x 256 out channels.
// ---------------------------------------------------------------------------
__launch_bounds__(512, 1)
__global__ void csc_wmma(const float* __restrict__ I,
                         const __bf16* __restrict__ Bp,
                         const float* __restrict__ w_b,
                         float* __restrict__ out) {
    extern __shared__ __bf16 Asm[];            // [MT][LDS_K] bf16, ~130 KB

    const int tid = threadIdx.x;
    const int w0  = blockIdx.x * TILE_W;
    const int h0  = blockIdx.y * TILE_H;
    const int b   = blockIdx.z;

    // ---------------- Phase 1: build A = [center | box] tile in LDS --------
    // 1024 tasks (c in 0..255, r in 0..3), 2 per thread. Sliding-window 3x3
    // box filter in fp32, cvt to bf16, scatter into the WMMA K dimension.
    #pragma unroll
    for (int t = 0; t < 2; ++t) {
        const int task = tid + t * 512;
        const int c = task & 255;
        const int r = task >> 8;                       // output row in tile
        const float* base = I + (size_t)(b * CIN + c) * (H_ * W_);
        const int ih0 = h0 + r;                        // input rows ih0..ih0+2
        const float* p0 = base + min(ih0 + 0, H_ - 1) * W_;
        const float* p1 = base + min(ih0 + 1, H_ - 1) * W_;
        const float* p2 = base + min(ih0 + 2, H_ - 1) * W_;

        float cs0, cs1, cs2, m1, m2;                   // column-sum window
        {
            int ca = min(w0 + 0, W_ - 1);
            int cb = min(w0 + 1, W_ - 1);
            float b0 = p1[ca], b1 = p1[cb];
            cs0 = p0[ca] + b0 + p2[ca];
            cs1 = p0[cb] + b1 + p2[cb];
            m1  = b1;                                  // I[ih0+1][w0+1]
        }
        for (int wv = 0; wv < TILE_W; ++wv) {
            int cx = min(w0 + wv + 2, W_ - 1);
            float bmid = p1[cx];
            cs2 = p0[cx] + bmid + p2[cx];
            m2  = bmid;
            float center = m1;                         // I[oh+1][ow+1]
            float box    = cs0 + cs1 + cs2;            // 3x3 box sum
            __bf16* dst = Asm + (size_t)(r * TILE_W + wv) * LDS_K;
            dst[c]       = (__bf16)center;             // K row c     -> w_c-w_s
            dst[CIN + c] = (__bf16)box;                // K row 256+c -> w_s
            cs0 = cs1; cs1 = cs2; m1 = m2;
        }
    }
    __syncthreads();

    // ---------------- Phase 2: WMMA GEMM (K=512, bf16 x bf16 -> f32) -------
    const int wave = tid >> 5;
    const int lane = tid & 31;
    const int half = lane >> 4;        // which 16-lane half of the wave
    const int ln   = lane & 15;
    const int n    = wave * 16 + ln;   // output channel for this lane column

    const float bias = w_b[n];
    v8f acc[8];
    #pragma unroll
    for (int ms = 0; ms < 8; ++ms)
        #pragma unroll
        for (int i = 0; i < 8; ++i) acc[ms][i] = bias;

    const uint4v* BpV = (const uint4v*)Bp;   // 16 bf16 = 2 x uint4v per (kblk,n)

    for (int kb = 0; kb < 16; ++kb) {        // K blocks of 32
        // B fragment: lane holds K = 32*kb + 16*half + e, column n. One load
        // per kb shared across all 8 M-subtiles (L2-resident weights).
        v16bf bf;
        {
            const uint4v* pb = BpV + (size_t)((2 * kb + half) * COUT + n) * 2;
            ((uint4v*)&bf)[0] = pb[0];
            ((uint4v*)&bf)[1] = pb[1];
        }
        #pragma unroll
        for (int ms = 0; ms < 8; ++ms) {
            // A fragment: lane row m = ms*16+ln,
            // elems 0..15 = K[32kb+16h .. +15] (matches B fragment K order)
            const int m = ms * 16 + ln;
            const __bf16* ap = Asm + (size_t)m * LDS_K + kb * 32 + half * 16;
            v16bf af;
            ((uint4v*)&af)[0] = *(const uint4v*)(ap);        // ds_load_b128
            ((uint4v*)&af)[1] = *(const uint4v*)(ap + 8);    // ds_load_b128
            acc[ms] = __builtin_amdgcn_wmma_f32_16x16x32_bf16(
                false, af, false, bf, (short)0, acc[ms], false, false);
        }
    }

    // ---------------- Epilogue -----------------------------------------
    // C layout: VGPR r -> M = r + 8*half, N = lane&15. A lane's 8 values per
    // subtile are 8 consecutive ow at an 8-float-aligned base within one row
    // (oh constant). Pair into v2f (global_store_b64, 8B-aligned since the
    // row offset is always even) with a non-temporal hint: the output is
    // write-once/never-read -> keep L2 for weights + input halo instead.
    #pragma unroll
    for (int ms = 0; ms < 8; ++ms) {
        const int mbase = ms * 16 + 8 * half;          // lane's first M row
        const int oh  = h0 + (mbase >> 5);
        const int owb = w0 + (mbase & 31);             // multiple of 8
        if (oh < HO) {
            float* rowp = out + (((size_t)b * COUT + n) * HO + oh) * WO;
            #pragma unroll
            for (int p = 0; p < 4; ++p) {
                const int ow = owb + 2 * p;            // even
                if (ow < WO) {                         // ow<126 => ow+1<=125
                    v2f v2 = { acc[ms][2 * p], acc[ms][2 * p + 1] };
                    __builtin_nontemporal_store(v2, (v2f*)(rowp + ow));
                }
            }
        }
    }
}

extern "C" void kernel_launch(void* const* d_in, const int* in_sizes, int n_in,
                              void* d_out, int out_size, void* d_ws, size_t ws_size,
                              hipStream_t stream) {
    const float* I   = (const float*)d_in[0];
    const float* w_c = (const float*)d_in[1];
    const float* w_s = (const float*)d_in[2];
    const float* w_b = (const float*)d_in[3];
    float* out = (float*)d_out;
    __bf16* Bp = (__bf16*)d_ws;              // 512*256*2 = 256 KB workspace

    const size_t shmem = (size_t)MT * LDS_K * sizeof(__bf16);  // 133120 B
    (void)hipFuncSetAttribute((const void*)csc_wmma,
                              hipFuncAttributeMaxDynamicSharedMemorySize,
                              (int)shmem);

    prep_weights<<<dim3(512), dim3(256), 0, stream>>>(w_c, w_s, Bp);

    dim3 grid((WO + TILE_W - 1) / TILE_W,    // 4
              (HO + TILE_H - 1) / TILE_H,    // 32
              B_);                           // 16
    csc_wmma<<<grid, dim3(512), shmem, stream>>>(I, Bp, w_b, out);
}
